// GRCNN_65652870087589
// MI455X (gfx1250) — matrix operand order repaired and checked by reading, not compiled, not run on hardware
//
#include <hip/hip_runtime.h>

// ---------------------------------------------------------------------------
// Types for CDNA5 WMMA (wave32)
// ---------------------------------------------------------------------------
typedef __bf16 bf16_t;
typedef bf16_t         v16bf  __attribute__((ext_vector_type(16)));
typedef float          v8f    __attribute__((ext_vector_type(8)));
typedef unsigned short u16x16 __attribute__((ext_vector_type(16)));
typedef unsigned short u16x8  __attribute__((ext_vector_type(8)));
typedef unsigned short u16x4  __attribute__((ext_vector_type(4)));

__device__ __forceinline__ unsigned short f32_to_bf16_rne(float f) {
    unsigned int u = __builtin_bit_cast(unsigned int, f);
    unsigned int r = u + 0x7FFFu + ((u >> 16) & 1u);
    return (unsigned short)(r >> 16);
}
__device__ __forceinline__ float bf16_to_f32(unsigned short s) {
    return __builtin_bit_cast(float, (unsigned int)s << 16);
}

// ---------------------------------------------------------------------------
// CDNA5 async global->LDS copy (ASYNCcnt-tracked, bypasses VGPRs).
// vdst = 32-bit LDS byte address (low 32 bits of the flat shared address,
// see ISA 10.2); vaddr = 64-bit global address.
// ---------------------------------------------------------------------------
__device__ __forceinline__ void async_copy_b128(const void* gsrc, const void* lds_dst) {
    unsigned laddr = (unsigned)(unsigned long long)lds_dst;
    asm volatile("global_load_async_to_lds_b128 %0, %1, off"
                 :: "v"(laddr), "v"(gsrc) : "memory");
}
__device__ __forceinline__ void wait_asynccnt0() {
    asm volatile("s_wait_asynccnt 0x0" ::: "memory");
}

// ---------------------------------------------------------------------------
// Tiled GEMM:  C[M,N] = [relu]( A[M,Kpad] * Bt[N,Kpad]^T + bias[N] )
//   A: fp32 or bf16 (a_is_bf16, block-uniform), row-major, lda = Kpad
//   Bt: bf16, pre-transposed [Npad][Kpad], zero-padded
//   C: optional fp32 (Cf) and/or bf16 (Cb) outputs
// Preconditions: M % 128 == 0, Kpad % 64 == 0, Npad >= roundup(N,64).
// Block: 256 thr = 8 waves; BM=128, BN=64, KT=64; double-buffered LDS with
// async global->LDS staging overlapped with 8 v_wmma_f32_16x16x32_bf16/tile.
// ---------------------------------------------------------------------------
#define BM 128
#define BN 64
#define KT 64
#define TPB 256
#define LSTR (KT + 8)   // LDS row stride in shorts (144B, 16B-aligned)

__global__ __launch_bounds__(TPB)
void gemm_wmma_v3(const void* __restrict__ Ap, int a_is_bf16, int lda,
                  const unsigned short* __restrict__ Bt, int ldbt,
                  const float* __restrict__ bias,
                  float* __restrict__ Cf, int ldcf,
                  unsigned short* __restrict__ Cb, int ldcb,
                  int M, int N, int relu) {
    __shared__ __align__(16) unsigned short lAs[2][BM][LSTR];
    __shared__ __align__(16) unsigned short lBt[2][BN][LSTR];

    const int Kpad = lda;
    const int bm0 = blockIdx.y * BM;
    const int bn0 = blockIdx.x * BN;

    const int tid  = threadIdx.x;
    const int lane = tid & 31;
    const int wave = tid >> 5;
    const int half = lane >> 4;
    const int l16  = lane & 15;

    const float* Af          = (const float*)Ap;
    const unsigned short* Ab = (const unsigned short*)Ap;

    v8f acc[4];
#pragma unroll
    for (int j = 0; j < 4; ++j) acc[j] = (v8f){0.f,0.f,0.f,0.f,0.f,0.f,0.f,0.f};

    // stage one k-tile into LDS buffer `buf`
    auto stage = [&](int buf, int k0) {
        if (a_is_bf16) {
#pragma unroll
            for (int i = 0; i < 4; ++i) {            // 1024 x b128 async copies
                const int s = tid + i * TPB;
                const int r = s >> 3, c8 = (s & 7) * 8;
                async_copy_b128(&Ab[(size_t)(bm0 + r) * lda + k0 + c8], &lAs[buf][r][c8]);
            }
        } else {
#pragma unroll
            for (int i = 0; i < 8; ++i) {            // fp32 -> bf16 convert path
                const int s = tid + i * TPB;
                const int r = s >> 4, c4 = (s & 15) * 4;
                const float4 v = *(const float4*)&Af[(size_t)(bm0 + r) * lda + k0 + c4];
                u16x4 p;
                p[0] = f32_to_bf16_rne(v.x); p[1] = f32_to_bf16_rne(v.y);
                p[2] = f32_to_bf16_rne(v.z); p[3] = f32_to_bf16_rne(v.w);
                *(u16x4*)&lAs[buf][r][c4] = p;
            }
        }
#pragma unroll
        for (int i = 0; i < 2; ++i) {                // B tile: 512 x b128 async
            const int s = tid + i * TPB;
            const int r = s >> 3, c8 = (s & 7) * 8;
            async_copy_b128(&Bt[(size_t)(bn0 + r) * ldbt + k0 + c8], &lBt[buf][r][c8]);
        }
    };

    const int nk = Kpad / KT;
    stage(0, 0);
    wait_asynccnt0();
    __syncthreads();

    for (int kt = 0; kt < nk; ++kt) {
        const int cur = kt & 1;
        if (kt + 1 < nk) stage(cur ^ 1, (kt + 1) * KT);   // overlap copy w/ math

#pragma unroll
        for (int kk = 0; kk < KT; kk += 32) {
            // A frag 16x32 (CDNA5 layout): lanes0-15 M=lane K{0..7,16..23};
            // lanes16-31 M=lane-16 K{8..15,24..31}
            const int arow = wave * 16 + l16;
            const int klo  = kk + half * 8;
            u16x8 a0 = *(const u16x8*)&lAs[cur][arow][klo];
            u16x8 a1 = *(const u16x8*)&lAs[cur][arow][klo + 16];
            u16x16 au;
#pragma unroll
            for (int i = 0; i < 8; ++i) { au[i] = a0[i]; au[8 + i] = a1[i]; }
            const v16bf av = __builtin_bit_cast(v16bf, au);

            const int kb = kk + half * 8;
            v16bf bfr[4];
#pragma unroll
            for (int j = 0; j < 4; ++j) {            // load all B frags first
                const int bn = j * 16 + l16;
                u16x8 b0 = *(const u16x8*)&lBt[cur][bn][kb];
                u16x8 b1 = *(const u16x8*)&lBt[cur][bn][kb + 16];
                u16x16 bu;
#pragma unroll
                for (int i = 0; i < 8; ++i) { bu[i] = b0[i]; bu[8 + i] = b1[i]; }
                bfr[j] = __builtin_bit_cast(v16bf, bu);
            }
#pragma unroll
            for (int j = 0; j < 4; ++j)              // back-to-back WMMAs
                acc[j] = __builtin_amdgcn_wmma_f32_16x16x32_bf16(
                    false, av, false, bfr[j], (short)0, acc[j], false, false);
        }
        wait_asynccnt0();       // next tile landed in LDS
        __syncthreads();        // all waves done reading cur / writing nxt
    }

    // ---- epilogue: dual store fp32 / bf16, guard col < N only ----
    const int mhi = half * 8;
#pragma unroll
    for (int j = 0; j < 4; ++j) {
        const int col = bn0 + j * 16 + l16;
        if (col < N) {
            const float bv = bias ? bias[col] : 0.f;
#pragma unroll
            for (int r = 0; r < 8; ++r) {
                const int row = bm0 + wave * 16 + mhi + r;
                float v = acc[j][r] + bv;
                if (relu) v = fmaxf(v, 0.f);
                if (Cf) Cf[(size_t)row * ldcf + col] = v;
                if (Cb) Cb[(size_t)row * ldcb + col] = f32_to_bf16_rne(v);
            }
        }
    }
}

// ---------------------------------------------------------------------------
// Weight pre-pass: W[K,N] fp32 -> Bt[Npad,Kpad] bf16, zero-padded
// ---------------------------------------------------------------------------
__global__ void wt_transpose_pad(const float* __restrict__ W, unsigned short* __restrict__ Bt,
                                 int K, int N, int Kpad, int Npad) {
    int idx = blockIdx.x * blockDim.x + threadIdx.x;
    int total = Kpad * Npad;
    if (idx < total) {
        int n = idx / Kpad, k = idx - n * Kpad;
        float v = (k < K && n < N) ? W[(size_t)k * N + n] : 0.f;
        Bt[idx] = f32_to_bf16_rne(v);
    }
}

// ---------------------------------------------------------------------------
// Graph helper kernels
// ---------------------------------------------------------------------------
__global__ void deg_count_kernel(const int* __restrict__ rel, float* deg_s, float* deg_o, int n) {
    int i = blockIdx.x * blockDim.x + threadIdx.x;
    if (i < n) {
        atomicAdd(&deg_s[rel[2 * i + 0]], 1.0f);
        atomicAdd(&deg_o[rel[2 * i + 1]], 1.0f);
    }
}
__global__ void invdeg_kernel(float* d, int n) {
    int i = blockIdx.x * blockDim.x + threadIdx.x;
    if (i < n) d[i] = 1.0f / (d[i] + 1e-7f);
}

// acc[tgt] += fc[rel] * invdeg[tgt]
__global__ void scatter_norm_kernel(const float* __restrict__ fc, int cols,
                                    const int* __restrict__ rel, int which,
                                    const float* __restrict__ invdeg,
                                    float* __restrict__ acc) {
    const int r = blockIdx.x;
    const int tgt = rel[2 * r + which];
    const float w = invdeg[tgt];
    const float* src = fc + (size_t)r * cols;
    float* dst = acc + (size_t)tgt * cols;
    for (int c = threadIdx.x; c < cols; c += blockDim.x)
        atomicAdd(&dst[c], src[c] * w);
}

// y = x + 0.5*a (2D, optional bf16 shadow)
__global__ void axpy2d_kernel(const float* __restrict__ x, int ldx,
                              const float* __restrict__ a, int ldda,
                              float* __restrict__ y, int ldy,
                              unsigned short* __restrict__ yb, int ldyb, int cols) {
    const int r = blockIdx.y;
    for (int c = blockIdx.x * blockDim.x + threadIdx.x; c < cols; c += gridDim.x * blockDim.x) {
        float v = x[(size_t)r * ldx + c] + 0.5f * a[(size_t)r * ldda + c];
        y[(size_t)r * ldy + c] = v;
        if (yb) yb[(size_t)r * ldyb + c] = f32_to_bf16_rne(v);
    }
}

// p_out[rel] = p_in[rel] + 0.5*nrm*(fcs[subj] + fco[obj]); bf16 and/or fp32 in/out
__global__ void gather_update_v2(const float* __restrict__ pin_f,
                                 const unsigned short* __restrict__ pin_b, int ldp,
                                 const float* __restrict__ fcs,
                                 const float* __restrict__ fco, int ldfc,
                                 const int* __restrict__ rel,
                                 float* __restrict__ pout_f,
                                 unsigned short* __restrict__ pout_b, int cols) {
    const int r = blockIdx.x;
    const int s = rel[2 * r + 0];
    const int o = rel[2 * r + 1];
    const float nrm = 1.0f / (1.0f + 1e-7f);
    for (int c = threadIdx.x; c < cols; c += blockDim.x) {
        float base = pin_f ? pin_f[(size_t)r * ldp + c] : bf16_to_f32(pin_b[(size_t)r * ldp + c]);
        float v = base + 0.5f * nrm * (fcs[(size_t)s * ldfc + c] + fco[(size_t)o * ldfc + c]);
        if (pout_f) pout_f[(size_t)r * ldp + c] = v;
        pout_b[(size_t)r * ldp + c] = f32_to_bf16_rne(v);
    }
}

// strided [rows,ld] -> contiguous [rows,cols]
__global__ void pack2d_kernel(const float* __restrict__ src, int lds_,
                              float* __restrict__ dst, int cols) {
    const int r = blockIdx.y;
    for (int c = blockIdx.x * blockDim.x + threadIdx.x; c < cols; c += gridDim.x * blockDim.x)
        dst[(size_t)r * cols + c] = src[(size_t)r * lds_ + c];
}

// ---------------------------------------------------------------------------
// Host orchestration
// ---------------------------------------------------------------------------
static inline void launch_gemm(const void* A, int a_bf16, int lda,
                               const unsigned short* Bt, int ldbt, const float* bias,
                               float* Cf, int ldcf, unsigned short* Cb, int ldcb,
                               int M, int N, int relu, hipStream_t s) {
    dim3 grid((N + BN - 1) / BN, M / BM);
    gemm_wmma_v3<<<grid, TPB, 0, s>>>(A, a_bf16, lda, Bt, ldbt, bias, Cf, ldcf, Cb, ldcb, M, N, relu);
}

extern "C" void kernel_launch(void* const* d_in, const int* in_sizes, int n_in,
                              void* d_out, int out_size, void* d_ws, size_t ws_size,
                              hipStream_t stream) {
    (void)in_sizes; (void)n_in; (void)out_size; (void)ws_size;

    constexpr int N_OBJ = 2048, N_REL = 32768, FEAT = 2048;
    constexpr int DIM = 512, NC_OBJ = 151, NC_PRED = 51;
    constexpr int KP_OS = 192;  // pad(151) to mult of 64
    constexpr int KP_PS = 64;   // pad(51)

    const float* x_obj_raw  = (const float*)d_in[0];
    const float* x_pred_raw = (const float*)d_in[1];
    const int*   rel_inds   = (const int*)  d_in[2];
    const float* obj_W1 = (const float*)d_in[3];  const float* obj_b1 = (const float*)d_in[4];
    const float* obj_W2 = (const float*)d_in[5];  const float* obj_b2 = (const float*)d_in[6];
    const float* rel_W1 = (const float*)d_in[7];  const float* rel_b1 = (const float*)d_in[8];
    const float* rel_W2 = (const float*)d_in[9];  const float* rel_b2 = (const float*)d_in[10];
    const float* Wc_feat    = (const float*)d_in[11];
    const float* bc_feat    = (const float*)d_in[12];
    const float* Wc_score_o = (const float*)d_in[13];
    const float* bc_score_o = (const float*)d_in[14];
    const float* Wc_score_r = (const float*)d_in[15];
    const float* bc_score_r = (const float*)d_in[16];
    const float* W_obj_cls  = (const float*)d_in[17]; const float* b_obj_cls  = (const float*)d_in[18];
    const float* W_pred_cls = (const float*)d_in[19]; const float* b_pred_cls = (const float*)d_in[20];

    float* out = (float*)d_out;

    constexpr size_t SZ_PF = (size_t)N_REL * DIM;
    constexpr size_t SZ_OS = (size_t)N_OBJ * NC_OBJ;
    constexpr size_t SZ_PS = (size_t)N_REL * NC_PRED;

    // ---- workspace layout (bump allocator, 256B aligned) ----
    char* wsp = (char*)d_ws;
    size_t off = 0;
    auto alloc = [&](size_t bytes) -> char* {
        char* p = wsp + off;
        off += (bytes + 255) & ~(size_t)255;
        return p;
    };
    typedef unsigned short u16;
    u16* PB0  = (u16*)alloc((size_t)N_REL * DIM * 2);   // pfeat bf16 ping
    u16* PB1  = (u16*)alloc((size_t)N_REL * DIM * 2);   // pong; also h_pred
    u16* OFB0 = (u16*)alloc((size_t)N_OBJ * DIM * 2);
    u16* OFB1 = (u16*)alloc((size_t)N_OBJ * DIM * 2);   // also h_obj
    u16* OSB0 = (u16*)alloc((size_t)N_OBJ * KP_OS * 2);
    u16* OSB1 = (u16*)alloc((size_t)N_OBJ * KP_OS * 2);
    u16* PSB0 = (u16*)alloc((size_t)N_REL * KP_PS * 2);
    u16* PSB1 = (u16*)alloc((size_t)N_REL * KP_PS * 2);
    // transposed/padded bf16 weights
    u16* WT_objW1   = (u16*)alloc((size_t)512 * 2048 * 2);
    u16* WT_objW2   = (u16*)alloc((size_t)512 * 512 * 2);
    u16* WT_relW1   = (u16*)alloc((size_t)512 * 2048 * 2);
    u16* WT_relW2   = (u16*)alloc((size_t)512 * 512 * 2);
    u16* WT_feat[4];
    for (int i = 0; i < 4; ++i) WT_feat[i] = (u16*)alloc((size_t)512 * 512 * 2);
    u16* WT_objcls  = (u16*)alloc((size_t)KP_OS * 512 * 2);
    u16* WT_predcls = (u16*)alloc((size_t)KP_PS * 512 * 2);
    u16* WT_so[2];  for (int i = 0; i < 2; ++i) WT_so[i] = (u16*)alloc((size_t)KP_OS * KP_PS * 2);
    u16* WT_sr[2];  for (int i = 0; i < 2; ++i) WT_sr[i] = (u16*)alloc((size_t)KP_PS * KP_OS * 2);
    // fp32 buffers
    float* FC_P   = (float*)alloc(SZ_PF * 4);           // also FC_PS [N_REL,151]
    float* OFE0   = (float*)alloc((size_t)N_OBJ * DIM * 4);
    float* OFE1   = (float*)alloc((size_t)N_OBJ * DIM * 4);
    float* FC_O   = (float*)alloc((size_t)N_OBJ * DIM * 4);
    float* FC_O2  = (float*)alloc((size_t)N_OBJ * DIM * 4);
    float* ACC_O  = (float*)alloc((size_t)N_OBJ * DIM * 4);
    float* OS0    = (float*)alloc((size_t)N_OBJ * KP_OS * 4);
    float* OS1    = (float*)alloc((size_t)N_OBJ * KP_OS * 4);
    float* ACC_OS = (float*)alloc((size_t)N_OBJ * NC_OBJ * 4);
    float* PS0    = (float*)alloc((size_t)N_REL * KP_PS * 4);
    float* PS1    = (float*)alloc((size_t)N_REL * KP_PS * 4);
    float* DEG_S  = (float*)alloc((size_t)N_OBJ * 4);
    float* DEG_O  = (float*)alloc((size_t)N_OBJ * 4);

    // ---- zero padded score buffers (K-pad columns must stay 0) ----
    hipMemsetAsync(OSB0, 0, (size_t)N_OBJ * KP_OS * 2 * 2, stream); // OSB0+OSB1
    hipMemsetAsync(PSB0, 0, (size_t)N_REL * KP_PS * 2 * 2, stream); // PSB0+PSB1
    hipMemsetAsync(OS0,  0, (size_t)N_OBJ * KP_OS * 4 * 2, stream); // OS0+OS1
    hipMemsetAsync(PS0,  0, (size_t)N_REL * KP_PS * 4 * 2, stream); // PS0+PS1
    hipMemsetAsync(DEG_S, 0, 2 * N_OBJ * 4, stream);

    // ---- degrees ----
    deg_count_kernel<<<(N_REL + 255) / 256, 256, 0, stream>>>(rel_inds, DEG_S, DEG_O, N_REL);
    invdeg_kernel<<<(2 * N_OBJ + 255) / 256, 256, 0, stream>>>(DEG_S, 2 * N_OBJ);

    // ---- weight pre-pass: transpose + pad + bf16 ----
    auto wt = [&](const float* W, u16* Bt, int K, int N, int Kpad, int Npad) {
        int tot = Kpad * Npad;
        wt_transpose_pad<<<(tot + 255) / 256, 256, 0, stream>>>(W, Bt, K, N, Kpad, Npad);
    };
    wt(obj_W1, WT_objW1, FEAT, DIM, FEAT, DIM);
    wt(obj_W2, WT_objW2, DIM, DIM, DIM, DIM);
    wt(rel_W1, WT_relW1, FEAT, DIM, FEAT, DIM);
    wt(rel_W2, WT_relW2, DIM, DIM, DIM, DIM);
    for (int i = 0; i < 4; ++i) wt(Wc_feat + (size_t)i * DIM * DIM, WT_feat[i], DIM, DIM, DIM, DIM);
    wt(W_obj_cls, WT_objcls, DIM, NC_OBJ, DIM, KP_OS);
    wt(W_pred_cls, WT_predcls, DIM, NC_PRED, DIM, KP_PS);
    for (int i = 0; i < 2; ++i) wt(Wc_score_o + (size_t)i * NC_PRED * NC_OBJ, WT_so[i], NC_PRED, NC_OBJ, KP_PS, KP_OS);
    for (int i = 0; i < 2; ++i) wt(Wc_score_r + (size_t)i * NC_OBJ * NC_PRED, WT_sr[i], NC_OBJ, NC_PRED, KP_OS, KP_PS);

    // ---- embedding MLPs ----
    launch_gemm(x_obj_raw, 0, FEAT, WT_objW1, FEAT, obj_b1, nullptr, 0, OFB1, DIM, N_OBJ, DIM, 1, stream);
    launch_gemm(OFB1, 1, DIM, WT_objW2, DIM, obj_b2, OFE0, DIM, OFB0, DIM, N_OBJ, DIM, 0, stream);
    launch_gemm(x_pred_raw, 0, FEAT, WT_relW1, FEAT, rel_b1, nullptr, 0, PB1, DIM, N_REL, DIM, 1, stream);
    launch_gemm(PB1, 1, DIM, WT_relW2, DIM, rel_b2, out, DIM, PB0, DIM, N_REL, DIM, 0, stream); // x_pred

    // ---- feature-level aGCN (2 steps) ----
    for (int s = 0; s < 2; ++s) {
        u16* pb_in  = s ? PB1 : PB0;   u16* pb_out = s ? PB0 : PB1;
        float* oe_in = s ? OFE1 : OFE0; float* oe_out = s ? OFE0 : OFE1;
        u16* ob_in  = s ? OFB1 : OFB0; u16* ob_out = s ? OFB0 : OFB1;

        hipMemsetAsync(ACC_O, 0, (size_t)N_OBJ * DIM * 4, stream);
        launch_gemm(pb_in, 1, DIM, WT_feat[0], DIM, bc_feat + 0 * DIM, FC_P, DIM, nullptr, 0, N_REL, DIM, 1, stream);
        scatter_norm_kernel<<<N_REL, 128, 0, stream>>>(FC_P, DIM, rel_inds, 0, DEG_S, ACC_O);
        launch_gemm(pb_in, 1, DIM, WT_feat[1], DIM, bc_feat + 1 * DIM, FC_P, DIM, nullptr, 0, N_REL, DIM, 1, stream);
        scatter_norm_kernel<<<N_REL, 128, 0, stream>>>(FC_P, DIM, rel_inds, 1, DEG_O, ACC_O);
        axpy2d_kernel<<<dim3(2, N_OBJ), 256, 0, stream>>>(oe_in, DIM, ACC_O, DIM, oe_out, DIM, ob_out, DIM, DIM);

        launch_gemm(ob_in, 1, DIM, WT_feat[2], DIM, bc_feat + 2 * DIM, FC_O, DIM, nullptr, 0, N_OBJ, DIM, 1, stream);
        launch_gemm(ob_in, 1, DIM, WT_feat[3], DIM, bc_feat + 3 * DIM, FC_O2, DIM, nullptr, 0, N_OBJ, DIM, 1, stream);
        gather_update_v2<<<N_REL, 128, 0, stream>>>(nullptr, pb_in, DIM, FC_O, FC_O2, DIM,
                                                    rel_inds, nullptr, pb_out, DIM);
    }

    // ---- classifiers (final: ofeat=OFE0/OFB0, pfeat=PB0) ----
    launch_gemm(OFB0, 1, DIM, WT_objcls, DIM, b_obj_cls, OS0, KP_OS, OSB0, KP_OS, N_OBJ, NC_OBJ, 0, stream);
    launch_gemm(PB0, 1, DIM, WT_predcls, DIM, b_pred_cls, PS0, KP_PS, PSB0, KP_PS, N_REL, NC_PRED, 0, stream);

    // ---- score-level aGCN (2 steps) ----
    for (int s = 0; s < 2; ++s) {
        float* ps_in = s ? PS1 : PS0;  float* ps_out = s ? PS0 : PS1;
        u16* psb_in  = s ? PSB1 : PSB0; u16* psb_out = s ? PSB0 : PSB1;
        float* os_in = s ? OS1 : OS0;  float* os_out = s ? OS0 : OS1;
        u16* osb_in  = s ? OSB1 : OSB0; u16* osb_out = s ? OSB0 : OSB1;

        hipMemsetAsync(ACC_OS, 0, (size_t)N_OBJ * NC_OBJ * 4, stream);
        launch_gemm(psb_in, 1, KP_PS, WT_so[0], KP_PS, bc_score_o + 0, FC_P, NC_OBJ, nullptr, 0, N_REL, NC_OBJ, 1, stream);
        scatter_norm_kernel<<<N_REL, 64, 0, stream>>>(FC_P, NC_OBJ, rel_inds, 0, DEG_S, ACC_OS);
        launch_gemm(psb_in, 1, KP_PS, WT_so[1], KP_PS, bc_score_o + NC_OBJ, FC_P, NC_OBJ, nullptr, 0, N_REL, NC_OBJ, 1, stream);
        scatter_norm_kernel<<<N_REL, 64, 0, stream>>>(FC_P, NC_OBJ, rel_inds, 1, DEG_O, ACC_OS);
        axpy2d_kernel<<<dim3(1, N_OBJ), 256, 0, stream>>>(os_in, KP_OS, ACC_OS, NC_OBJ, os_out, KP_OS, osb_out, KP_OS, NC_OBJ);

        launch_gemm(osb_in, 1, KP_OS, WT_sr[0], KP_OS, bc_score_r + 0, FC_O, NC_PRED, nullptr, 0, N_OBJ, NC_PRED, 1, stream);
        launch_gemm(osb_in, 1, KP_OS, WT_sr[1], KP_OS, bc_score_r + NC_PRED, FC_O2, NC_PRED, nullptr, 0, N_OBJ, NC_PRED, 1, stream);
        gather_update_v2<<<N_REL, 64, 0, stream>>>(ps_in, psb_in, KP_PS, FC_O, FC_O2, NC_PRED,
                                                   rel_inds, ps_out, psb_out, NC_PRED);
    }

    // ---- pack outputs: x_pred already at out[0]; os_, ps_ strided -> packed ----
    pack2d_kernel<<<dim3(1, N_OBJ), 256, 0, stream>>>(OS0, KP_OS, out + SZ_PF, NC_OBJ);
    pack2d_kernel<<<dim3(1, N_REL), 64, 0, stream>>>(PS0, KP_PS, out + SZ_PF + SZ_OS, NC_PRED);
}
